// GIN_80075370267117
// MI455X (gfx1250) — compile-verified, matrix operand order checked by reading, not run on hardware
//
#include <hip/hip_runtime.h>
#include <stdint.h>

typedef __attribute__((ext_vector_type(16))) _Float16 v16h;
typedef __attribute__((ext_vector_type(8)))  float    v8f;

union FragU { v16h v; uint32_t u[8]; };

#define CIN  128
#define HID  256
#define COUT 128
#define NG   512
#define BN_EPS 1e-5f

#define WMMA_CALL(A_, B_, C_) \
  __builtin_amdgcn_wmma_f32_16x16x32_f16(false, (A_).v, false, (B_).v, (short)0, (C_), false, false)

// ---------------- elementwise / conversion kernels ----------------

__global__ void k_cvt_f16(const float* __restrict__ in, _Float16* __restrict__ out, size_t n) {
  size_t i = (size_t)blockIdx.x * blockDim.x + threadIdx.x;
  if (i < n) out[i] = (_Float16)in[i];
}

__global__ void k_copy_f32(const float* __restrict__ in, float* __restrict__ out, size_t n) {
  size_t i = (size_t)blockIdx.x * blockDim.x + threadIdx.x;
  if (i < n) out[i] = in[i];
}

__global__ void k_zero_f32(float* __restrict__ p, size_t n) {
  size_t i = (size_t)blockIdx.x * blockDim.x + threadIdx.x;
  if (i < n) p[i] = 0.0f;
}

// W: K x Nc (f32, row-major)  ->  Wt: Nc x K (f16, row-major) == B^T
__global__ void k_transpose_f16(const float* __restrict__ W, _Float16* __restrict__ Wt,
                                int K, int Nc) {
  int idx = blockIdx.x * blockDim.x + threadIdx.x;
  if (idx >= K * Nc) return;
  int k = idx / Nc, n = idx % Nc;
  Wt[(size_t)n * K + k] = (_Float16)W[idx];
}

// ---------------- graph scatter kernels ----------------

// one block per edge, blockDim = C; agg[dst] += X[src]  (agg pre-initialized to X)
__global__ void k_edge_scatter(const float* __restrict__ X, const int* __restrict__ src,
                               const int* __restrict__ dst, float* __restrict__ agg, int C) {
  int e = blockIdx.x;
  int s = src[e], d = dst[e];
  atomicAdd(&agg[(size_t)d * C + threadIdx.x], X[(size_t)s * C + threadIdx.x]);
}

// one block per node, blockDim = C
__global__ void k_pool(const float* __restrict__ H, const int* __restrict__ batch,
                       float* __restrict__ pooled, int C) {
  int node = blockIdx.x;
  int b = batch[node];
  atomicAdd(&pooled[(size_t)b * C + threadIdx.x], H[(size_t)node * C + threadIdx.x]);
}

// ---------------- WMMA GEMM ----------------
// C[M x Nc] = act(A[M x K](f16) * B[K x Nc] + bias), B given as Bt[Nc x K] (f16).
// One wave -> 32x64 output strip (2 row-tiles x 4 col-tiles, 8 accumulators).
// Block = 256 threads = 8 waves: 4 in M (128 rows) x 2 in N (128 cols).
// A buffers must be padded to >= (row-tiles rounded up to 32 rows).
template <int RELU, int OUT_F16>
__global__ void k_wmma_gemm(const _Float16* __restrict__ A,
                            const _Float16* __restrict__ Bt,
                            const float* __restrict__ bias,
                            float* __restrict__ Cf,
                            _Float16* __restrict__ Ch,
                            int M, int K, int Nc) {
  const int lane  = threadIdx.x & 31;
  const int wave  = threadIdx.x >> 5;
  const int waveM = wave & 3;
  const int waveN = wave >> 2;
  const int row0  = blockIdx.x * 128 + waveM * 32;
  const int col0  = blockIdx.y * 128 + waveN * 64;
  if (row0 >= M) return;  // wave-uniform: EXEC stays all-ones for WMMA

  const int mn = lane & 15;   // tile-local row (A) / col (C, Bt)
  const int hg = lane >> 4;   // lane half-group
  const _Float16* __restrict__ ar0 = A + (size_t)(row0 + mn) * K;
  const _Float16* __restrict__ ar1 = ar0 + (size_t)16 * K;
  const _Float16* __restrict__ br  = Bt + (size_t)(col0 + mn) * K;
  const size_t bstep = (size_t)16 * K;

  v8f acc[2][4] = {};
  for (int k0 = 0; k0 < K; k0 += 32) {
    FragU a0, a1, b0, b1, b2, b3;
#pragma unroll
    for (int i = 0; i < 8; ++i) {
      // 16-bit fragment layout (ISA 7.12.2): dword i holds K = base, base+1
      const int kb = k0 + ((i & 4) ? 16 : 0) + hg * 8 + ((i & 3) << 1);
      a0.u[i] = *(const uint32_t*)(ar0 + kb);
      a1.u[i] = *(const uint32_t*)(ar1 + kb);
      b0.u[i] = *(const uint32_t*)(br + kb);
      b1.u[i] = *(const uint32_t*)(br + bstep + kb);
      b2.u[i] = *(const uint32_t*)(br + 2 * bstep + kb);
      b3.u[i] = *(const uint32_t*)(br + 3 * bstep + kb);
    }
    acc[0][0] = WMMA_CALL(a0, b0, acc[0][0]);
    acc[0][1] = WMMA_CALL(a0, b1, acc[0][1]);
    acc[0][2] = WMMA_CALL(a0, b2, acc[0][2]);
    acc[0][3] = WMMA_CALL(a0, b3, acc[0][3]);
    acc[1][0] = WMMA_CALL(a1, b0, acc[1][0]);
    acc[1][1] = WMMA_CALL(a1, b1, acc[1][1]);
    acc[1][2] = WMMA_CALL(a1, b2, acc[1][2]);
    acc[1][3] = WMMA_CALL(a1, b3, acc[1][3]);
  }

#pragma unroll
  for (int rt = 0; rt < 2; ++rt) {
    if (row0 + rt * 16 < M) {  // wave-uniform guard (M % 16 == 0)
#pragma unroll
      for (int ct = 0; ct < 4; ++ct) {
        const float bv = bias[col0 + ct * 16 + mn];
#pragma unroll
        for (int r = 0; r < 8; ++r) {
          float v = acc[rt][ct][r] + bv;
          if (RELU) v = fmaxf(v, 0.0f);
          const size_t idx =
              (size_t)(row0 + rt * 16 + r + hg * 8) * Nc + (col0 + ct * 16 + mn);
          if (OUT_F16) Ch[idx] = (_Float16)v;
          else         Cf[idx] = v;
        }
      }
    }
  }
}

// ---------------- BatchNorm pieces ----------------

// blockDim == C; each block reduces a contiguous chunk of rows (coalesced),
// then atomically accumulates per-column sum / sumsq.
__global__ void k_colsum_partial(const float* __restrict__ X, int Nrows, int C,
                                 int rowsPerBlock,
                                 float* __restrict__ sums, float* __restrict__ sqs) {
  const int c  = threadIdx.x;
  int r0 = blockIdx.x * rowsPerBlock;
  int r1 = r0 + rowsPerBlock;
  if (r1 > Nrows) r1 = Nrows;
  float s = 0.0f, s2 = 0.0f;
  for (int r = r0; r < r1; ++r) {
    float v = X[(size_t)r * C + c];
    s += v; s2 += v * v;
  }
  atomicAdd(&sums[c], s);
  atomicAdd(&sqs[c], s2);
}

// h = relu(res + gamma*(hb-mean)*rsqrt(var+eps)+beta); mean/var from sums/sqs.
// Writes f32 (next scatter) and f16 (next GEMM) copies.
__global__ void k_bn_res_relu(const float* __restrict__ hb, const float* __restrict__ res,
                              const float* __restrict__ sums, const float* __restrict__ sqs,
                              const float* __restrict__ gamma, const float* __restrict__ beta,
                              float* __restrict__ outf, _Float16* __restrict__ outh,
                              size_t total, int C, float invN) {
  size_t i = (size_t)blockIdx.x * blockDim.x + threadIdx.x;
  if (i >= total) return;
  int c = (int)(i % (size_t)C);
  float mu  = sums[c] * invN;
  float var = sqs[c] * invN - mu * mu;
  float v = gamma[c] * (hb[i] - mu) * rsqrtf(var + BN_EPS) + beta[c] + res[i];
  v = fmaxf(v, 0.0f);
  outf[i] = v;
  outh[i] = (_Float16)v;
}

// ---------------- log-softmax ----------------
// one block per row, blockDim == C == 128
__global__ void k_logsoftmax(const float* __restrict__ X, float* __restrict__ out, int C) {
  const int row = blockIdx.x, t = threadIdx.x;
  __shared__ float sh[128];
  float v = X[(size_t)row * C + t];
  sh[t] = v; __syncthreads();
  for (int o = 64; o > 0; o >>= 1) { if (t < o) sh[t] = fmaxf(sh[t], sh[t + o]); __syncthreads(); }
  float mx = sh[0]; __syncthreads();
  sh[t] = __expf(v - mx); __syncthreads();
  for (int o = 64; o > 0; o >>= 1) { if (t < o) sh[t] += sh[t + o]; __syncthreads(); }
  float lse = mx + __logf(sh[0]);
  out[(size_t)row * C + t] = v - lse;
}

// ---------------- host orchestration ----------------

static inline size_t cdiv(size_t a, size_t b) { return (a + b - 1) / b; }

extern "C" void kernel_launch(void* const* d_in, const int* in_sizes, int n_in,
                              void* d_out, int out_size, void* d_ws, size_t ws_size,
                              hipStream_t stream) {
  const float* x    = (const float*)d_in[0];
  const int*   ei   = (const int*)  d_in[1];
  const int*   bidx = (const int*)  d_in[2];
  const float* w1a  = (const float*)d_in[3];
  const float* b1a  = (const float*)d_in[4];
  const float* w1b  = (const float*)d_in[5];
  const float* b1b  = (const float*)d_in[6];
  const float* w2a  = (const float*)d_in[7];
  const float* b2a  = (const float*)d_in[8];
  const float* w2b  = (const float*)d_in[9];
  const float* b2b  = (const float*)d_in[10];
  const float* g1   = (const float*)d_in[11];
  const float* be1  = (const float*)d_in[12];
  const float* g2   = (const float*)d_in[13];
  const float* be2  = (const float*)d_in[14];
  const float* wr1  = (const float*)d_in[15];
  const float* br1  = (const float*)d_in[16];
  const float* wr2  = (const float*)d_in[17];
  const float* br2  = (const float*)d_in[18];

  const size_t N  = (size_t)in_sizes[0] / CIN;
  const size_t E  = (size_t)in_sizes[1] / 2;
  const size_t Np = N + 32;  // padded rows for guarded 2nd row-tile loads
  const int*   src = ei;
  const int*   dst = ei + E;

  // workspace carve-out (256B aligned)
  char* ws = (char*)d_ws;
  auto alloc = [&](size_t bytes) -> char* {
    char* p = ws; ws += (bytes + 255) & ~(size_t)255; return p;
  };
  _Float16* xh    = (_Float16*)alloc(Np * CIN * 2);   // GEMM A operand -> padded
  float*    aggf  = (float*)   alloc(N  * HID * 4);
  _Float16* aggh  = (_Float16*)alloc(Np * HID * 2);   // padded
  _Float16* th    = (_Float16*)alloc(Np * HID * 2);   // padded; also f16 sink
  float*    hb    = (float*)   alloc(N  * HID * 4);
  float*    resf  = (float*)   alloc(N  * HID * 4);
  float*    h1f   = (float*)   alloc(N  * HID * 4);
  _Float16* h1h   = (_Float16*)alloc(Np * HID * 2);   // padded
  float*    h2f   = (float*)   alloc(N  * COUT * 4);
  float*    sumv  = (float*)   alloc(HID * 4);
  float*    sqv   = (float*)   alloc(HID * 4);
  float*    pooled= (float*)   alloc((size_t)NG * COUT * 4);
  _Float16* w1aT  = (_Float16*)alloc((size_t)HID * CIN * 2);
  _Float16* w1bT  = (_Float16*)alloc((size_t)HID * HID * 2);
  _Float16* wr1T  = (_Float16*)alloc((size_t)HID * CIN * 2);
  _Float16* w2aT  = (_Float16*)alloc((size_t)HID * HID * 2);
  _Float16* w2bT  = (_Float16*)alloc((size_t)COUT * HID * 2);
  _Float16* wr2T  = (_Float16*)alloc((size_t)COUT * HID * 2);

  const int TB = 256;
  const float invN = 1.0f / (float)N;

  // weights -> f16 transposed (B^T layout)
  k_transpose_f16<<<cdiv(CIN * HID, TB), TB, 0, stream>>>(w1a, w1aT, CIN, HID);
  k_transpose_f16<<<cdiv(HID * HID, TB), TB, 0, stream>>>(w1b, w1bT, HID, HID);
  k_transpose_f16<<<cdiv(CIN * HID, TB), TB, 0, stream>>>(wr1, wr1T, CIN, HID);
  k_transpose_f16<<<cdiv(HID * HID, TB), TB, 0, stream>>>(w2a, w2aT, HID, HID);
  k_transpose_f16<<<cdiv(HID * COUT, TB), TB, 0, stream>>>(w2b, w2bT, HID, COUT);
  k_transpose_f16<<<cdiv(HID * COUT, TB), TB, 0, stream>>>(wr2, wr2T, HID, COUT);

  const dim3 gemmBlk(256);
  const dim3 gemmGridH(cdiv(N, 128), HID / 128);   // Nc = 256 -> grid.y = 2
  const dim3 gemmGridC(cdiv(N, 128), COUT / 128);  // Nc = 128 -> grid.y = 1

  const int statBlocks = 512;
  const int rowsPerBlk = (int)cdiv(N, statBlocks);

  // ---------------- layer 1 ----------------
  k_cvt_f16 <<<cdiv(N * CIN, TB), TB, 0, stream>>>(x, xh, N * CIN);
  k_copy_f32<<<cdiv(N * CIN, TB), TB, 0, stream>>>(x, aggf, N * CIN);
  k_edge_scatter<<<(uint32_t)E, CIN, 0, stream>>>(x, src, dst, aggf, CIN);
  k_cvt_f16 <<<cdiv(N * CIN, TB), TB, 0, stream>>>(aggf, aggh, N * CIN);

  // t = relu((x+agg) @ w1a + b1a)  [f16 out]
  k_wmma_gemm<1, 1><<<gemmGridH, gemmBlk, 0, stream>>>(aggh, w1aT, b1a, nullptr, th,
                                                       (int)N, CIN, HID);
  // hb = t @ w1b + b1b  [f32 out]
  k_wmma_gemm<0, 0><<<gemmGridH, gemmBlk, 0, stream>>>(th, w1bT, b1b, hb, nullptr,
                                                       (int)N, HID, HID);
  // res = x @ wr1 + br1  [f32 out]
  k_wmma_gemm<0, 0><<<gemmGridH, gemmBlk, 0, stream>>>(xh, wr1T, br1, resf, nullptr,
                                                       (int)N, CIN, HID);
  k_zero_f32<<<1, HID, 0, stream>>>(sumv, HID);
  k_zero_f32<<<1, HID, 0, stream>>>(sqv, HID);
  k_colsum_partial<<<statBlocks, HID, 0, stream>>>(hb, (int)N, HID, rowsPerBlk, sumv, sqv);
  k_bn_res_relu<<<cdiv(N * HID, TB), TB, 0, stream>>>(hb, resf, sumv, sqv, g1, be1,
                                                      h1f, h1h, N * HID, HID, invN);

  // ---------------- layer 2 ----------------
  k_copy_f32<<<cdiv(N * HID, TB), TB, 0, stream>>>(h1f, aggf, N * HID);
  k_edge_scatter<<<(uint32_t)E, HID, 0, stream>>>(h1f, src, dst, aggf, HID);
  k_cvt_f16 <<<cdiv(N * HID, TB), TB, 0, stream>>>(aggf, aggh, N * HID);

  k_wmma_gemm<1, 1><<<gemmGridH, gemmBlk, 0, stream>>>(aggh, w2aT, b2a, nullptr, th,
                                                       (int)N, HID, HID);
  k_wmma_gemm<0, 0><<<gemmGridC, gemmBlk, 0, stream>>>(th, w2bT, b2b, hb, nullptr,
                                                       (int)N, HID, COUT);
  k_wmma_gemm<0, 0><<<gemmGridC, gemmBlk, 0, stream>>>(h1h, wr2T, br2, resf, nullptr,
                                                       (int)N, HID, COUT);
  k_zero_f32<<<1, COUT, 0, stream>>>(sumv, COUT);
  k_zero_f32<<<1, COUT, 0, stream>>>(sqv, COUT);
  k_colsum_partial<<<statBlocks, COUT, 0, stream>>>(hb, (int)N, COUT, rowsPerBlk, sumv, sqv);
  k_bn_res_relu<<<cdiv(N * COUT, TB), TB, 0, stream>>>(hb, resf, sumv, sqv, g2, be2,
                                                       h2f, th, N * COUT, COUT, invN);

  // ---------------- pooling + log-softmax ----------------
  k_zero_f32<<<cdiv((size_t)NG * COUT, TB), TB, 0, stream>>>(pooled, (size_t)NG * COUT);
  k_pool<<<(uint32_t)N, COUT, 0, stream>>>(h2f, bidx, pooled, COUT);
  k_logsoftmax<<<NG, COUT, 0, stream>>>(pooled, (float*)d_out, COUT);
}
